// OptimizedMultiAttentionHead_68436008894918
// MI455X (gfx1250) — matrix operand order repaired and checked by reading, not compiled, not run on hardware
//
#include <hip/hip_runtime.h>

// ---------------------------------------------------------------------------
// Problem constants (B=2, T=2048, C=1024, H=16, hs=64)
// ---------------------------------------------------------------------------
#define BATCH 2
#define SEQ   2048
#define CH    1024
#define NH    16
#define HS    64
#define BT    (BATCH * SEQ)        // 4096 tokens
#define BH    (BATCH * NH)         // 32 (batch,head) pairs

typedef __attribute__((ext_vector_type(16))) __bf16 v16bf;
typedef __attribute__((ext_vector_type(8)))  __bf16 v8bf;
typedef __attribute__((ext_vector_type(8)))  float  v8f;

// round-to-nearest-even f32 -> bf16
static __device__ inline __bf16 f2bf(float f) {
  unsigned u = __builtin_bit_cast(unsigned, f);
  unsigned r = u + 0x7fffu + ((u >> 16) & 1u);
  unsigned short h = (unsigned short)(r >> 16);
  return __builtin_bit_cast(__bf16, h);
}

// A-matrix 16x32 bf16 fragment from a row-major row pointer (row already
// selected as lane&15). Lanes 0-15 hold K {0..15}; lanes 16-31 hold
// K {16..31}  (two 16B chunks -> v16bf across 8 VGPRs).
static __device__ inline v16bf load_afrag(const __bf16* row, int lane) {
  const int base = (lane < 16) ? 0 : 16;
  union { v16bf v; v8bf h[2]; } u;
  u.h[0] = *(const v8bf*)(row + base);
  u.h[1] = *(const v8bf*)(row + 8 + base);
  return u.v;
}

static __device__ inline v8f wmma_bf16(v16bf a, v16bf b, v8f c) {
  return __builtin_amdgcn_wmma_f32_16x16x32_bf16(
      /*neg_a=*/false, a, /*neg_b=*/false, b,
      /*c_mod=*/(short)0, c, /*reuse_a=*/false, /*reuse_b=*/false);
}

// ---------------------------------------------------------------------------
// f32 -> bf16 streaming convert
// ---------------------------------------------------------------------------
__global__ void cvt_f32_bf16(const float* __restrict__ in,
                             __bf16* __restrict__ out, int n) {
  int i = blockIdx.x * blockDim.x + threadIdx.x;
  int stride = gridDim.x * blockDim.x;
  for (; i < n; i += stride) out[i] = f2bf(in[i]);
}

// ---------------------------------------------------------------------------
// QKV GEMM: qkv[m,n] = sum_k x[m,k] * w_qkv[n,k]   (M=4096, N=3072, K=1024)
// Block = 8 waves, block tile 256x64; wave tile 32x64 (2 A-frags x 4 B-frags
// -> 8 WMMAs per K-step). Scatters results into q (scaled by hs^-0.5) and
// k as (B,H,T,hs) bf16, and v TRANSPOSED as vT (B,H,hs,T) bf16 so the PV
// GEMM's B-fragments are contiguous loads.
// ---------------------------------------------------------------------------
__global__ __launch_bounds__(256) void qkv_gemm(
    const __bf16* __restrict__ xb, const __bf16* __restrict__ wb,
    __bf16* __restrict__ qb, __bf16* __restrict__ kb, __bf16* __restrict__ vT) {
  const int lane = threadIdx.x & 31;
  const int wave = threadIdx.x >> 5;
  const int m0 = blockIdx.y * 256 + wave * 32;
  const int n0 = blockIdx.x * 64;
  const int lrow = lane & 15;
  const int half = lane >> 4;
  const int rowBase = half * 8;

  const __bf16* arow0 = xb + (size_t)(m0 + lrow) * CH;
  const __bf16* arow1 = xb + (size_t)(m0 + 16 + lrow) * CH;
  v8f acc0[4] = {};
  v8f acc1[4] = {};
  for (int kk = 0; kk < CH; kk += 32) {
    v16bf a0 = load_afrag(arow0 + kk, lane);
    v16bf a1 = load_afrag(arow1 + kk, lane);
    __builtin_prefetch(arow0 + kk + 256, 0, 3);   // global_prefetch_b8, ~8 steps ahead
    __builtin_prefetch(arow1 + kk + 256, 0, 3);
#pragma unroll
    for (int nt = 0; nt < 4; ++nt) {
      const __bf16* brow = wb + (size_t)(n0 + nt * 16 + lrow) * CH + kk + half * 16;
      if (nt == 0) __builtin_prefetch(brow + 256, 0, 3);
      v16bf b = *(const v16bf*)brow;
      acc0[nt] = wmma_bf16(a0, b, acc0[nt]);
      acc1[nt] = wmma_bf16(a1, b, acc1[nt]);
    }
  }
#pragma unroll
  for (int g = 0; g < 2; ++g) {
#pragma unroll
    for (int nt = 0; nt < 4; ++nt) {
      int n = n0 + nt * 16 + lrow;
#pragma unroll
      for (int r = 0; r < 8; ++r) {
        int m = m0 + g * 16 + rowBase + r;
        int bidx = m >> 11, t = m & (SEQ - 1);
        float v = g ? acc1[nt][r] : acc0[nt][r];
        if (n < CH) {                     // Q (pre-scaled by 1/sqrt(hs) = 0.125)
          int h = n >> 6, d = n & 63;
          qb[(size_t)((bidx * NH + h) * SEQ + t) * HS + d] = f2bf(v * 0.125f);
        } else if (n < 2 * CH) {          // K
          int n2 = n - CH, h = n2 >> 6, d = n2 & 63;
          kb[(size_t)((bidx * NH + h) * SEQ + t) * HS + d] = f2bf(v);
        } else {                          // V transposed: vT[b,h,d,t]
          int n3 = n - 2 * CH, h = n3 >> 6, d = n3 & 63;
          vT[(size_t)((bidx * NH + h) * HS + d) * SEQ + t] = f2bf(v);
        }
      }
    }
  }
}

// ---------------------------------------------------------------------------
// Attention pass 1: exact causal row-max / row-sumexp via online recurrence.
// Grid: (BH, T/128). Wave owns 16 q-rows; loops k-tiles of 16 (2 WMMAs each,
// K = hs = 64 split as 2x32).
// ---------------------------------------------------------------------------
__global__ __launch_bounds__(256) void attn_stats(
    const __bf16* __restrict__ qb, const __bf16* __restrict__ kb,
    float* __restrict__ mbuf, float* __restrict__ lbuf) {
  const int lane = threadIdx.x & 31;
  const int wave = threadIdx.x >> 5;
  const int bh = blockIdx.x;
  const int q0 = blockIdx.y * 128 + wave * 16;
  const int lrow = lane & 15;
  const int half = lane >> 4;
  const int rowBase = half * 8;

  const __bf16* qrowp = qb + (size_t)(bh * SEQ + q0 + lrow) * HS;
  v16bf aq0 = load_afrag(qrowp, lane);
  v16bf aq1 = load_afrag(qrowp + 32, lane);

  float mrow[8], srow[8];
#pragma unroll
  for (int r = 0; r < 8; ++r) { mrow[r] = -1e30f; srow[r] = 0.f; }

  const int kEnd = q0 + 16;
  for (int k0 = 0; k0 < kEnd; k0 += 16) {
    const __bf16* krowp = kb + (size_t)(bh * SEQ + k0 + lrow) * HS + half * 16;
    __builtin_prefetch(krowp + 32 * HS, 0, 3);   // next two k-tiles ahead
    v16bf bk0 = *(const v16bf*)krowp;
    v16bf bk1 = *(const v16bf*)(krowp + 32);
    v8f s = {};
    s = wmma_bf16(aq0, bk0, s);
    s = wmma_bf16(aq1, bk1, s);
    int kcol = k0 + lrow;
#pragma unroll
    for (int r = 0; r < 8; ++r) {
      int qrow = q0 + rowBase + r;
      float sv = (kcol <= qrow) ? s[r] : -1e30f;
      float tmax = sv;
      tmax = fmaxf(tmax, __shfl_xor(tmax, 1, 32));
      tmax = fmaxf(tmax, __shfl_xor(tmax, 2, 32));
      tmax = fmaxf(tmax, __shfl_xor(tmax, 4, 32));
      tmax = fmaxf(tmax, __shfl_xor(tmax, 8, 32));
      float nm = fmaxf(mrow[r], tmax);
      float p = __expf(sv - nm);
      p += __shfl_xor(p, 1, 32);
      p += __shfl_xor(p, 2, 32);
      p += __shfl_xor(p, 4, 32);
      p += __shfl_xor(p, 8, 32);
      srow[r] = srow[r] * __expf(mrow[r] - nm) + p;
      mrow[r] = nm;
    }
  }
  if (lrow == 0) {
#pragma unroll
    for (int r = 0; r < 8; ++r) {
      mbuf[bh * SEQ + q0 + rowBase + r] = mrow[r];
      lbuf[bh * SEQ + q0 + rowBase + r] = srow[r];
    }
  }
}

// ---------------------------------------------------------------------------
// Attention pass 2: recompute S, exact softmax with final stats, apply the
// nonlinear cosine mask p*(0.5*cos(a1[k]*p+b1[k])+0.5), re-stage P through
// LDS (C-layout -> A-layout) as bf16, accumulate P'@V via WMMA with vT.
// Output stored bf16 in (B,T,C) layout ready for the projection GEMM.
// ---------------------------------------------------------------------------
__global__ __launch_bounds__(256) void attn_pv(
    const __bf16* __restrict__ qb, const __bf16* __restrict__ kb,
    const __bf16* __restrict__ vT, const float* __restrict__ mbuf,
    const float* __restrict__ lbuf, const float* __restrict__ a1,
    const float* __restrict__ b1, __bf16* __restrict__ attn_out) {
  __shared__ __align__(32) __bf16 pshm[8][16 * 32];   // per-wave 16x32 P tile
  const int lane = threadIdx.x & 31;
  const int wave = threadIdx.x >> 5;
  const int bh = blockIdx.x;
  const int bidx = bh >> 4;
  const int h = bh & 15;
  const int q0 = blockIdx.y * 128 + wave * 16;
  const int lrow = lane & 15;
  const int half = lane >> 4;
  const int rowBase = half * 8;

  const __bf16* qrowp = qb + (size_t)(bh * SEQ + q0 + lrow) * HS;
  v16bf aq0 = load_afrag(qrowp, lane);
  v16bf aq1 = load_afrag(qrowp + 32, lane);

  float mr[8], il[8];
#pragma unroll
  for (int r = 0; r < 8; ++r) {
    mr[r] = mbuf[bh * SEQ + q0 + rowBase + r];
    il[r] = 1.f / lbuf[bh * SEQ + q0 + rowBase + r];
  }

  v8f o[4] = {};
  const int kEnd = q0 + 16;
  for (int k0 = 0; k0 < kEnd; k0 += 32) {
#pragma unroll
    for (int t = 0; t < 2; ++t) {
      int kbase = k0 + 16 * t;
      const __bf16* krowp = kb + (size_t)(bh * SEQ + kbase + lrow) * HS + half * 16;
      if (t == 0) __builtin_prefetch(krowp + 32 * HS, 0, 3);
      v16bf bk0 = *(const v16bf*)krowp;
      v16bf bk1 = *(const v16bf*)(krowp + 32);
      v8f s = {};
      s = wmma_bf16(aq0, bk0, s);
      s = wmma_bf16(aq1, bk1, s);
      int kcol = kbase + lrow;
      float av = a1[kcol];
      float bv = b1[kcol];
#pragma unroll
      for (int r = 0; r < 8; ++r) {
        int qrow = q0 + rowBase + r;
        float p = (kcol <= qrow) ? __expf(s[r] - mr[r]) * il[r] : 0.f;
        p = p * (0.5f * __cosf(av * p + bv) + 0.5f);
        pshm[wave][(rowBase + r) * 32 + 16 * t + lrow] = f2bf(p);
      }
    }
    // Re-read P tile in A-fragment layout (same wave: DScnt wait inserted).
    v16bf ap = load_afrag(&pshm[wave][lrow * 32], lane);
#pragma unroll
    for (int dt = 0; dt < 4; ++dt) {
      const __bf16* vrow =
          vT + (size_t)(bh * HS + dt * 16 + lrow) * SEQ + k0 + half * 16;
      if (dt == 0) __builtin_prefetch(vrow + 32, 0, 3);
      v16bf bvv = *(const v16bf*)vrow;
      o[dt] = wmma_bf16(ap, bvv, o[dt]);
    }
  }
#pragma unroll
  for (int dt = 0; dt < 4; ++dt) {
    int c = h * HS + dt * 16 + lrow;
#pragma unroll
    for (int r = 0; r < 8; ++r) {
      int tok = bidx * SEQ + q0 + rowBase + r;
      attn_out[(size_t)tok * CH + c] = f2bf(o[dt][r]);
    }
  }
}

// ---------------------------------------------------------------------------
// Projection GEMM: out[m,c] = sum_k attn[m,k]*w_proj[c,k], then
// out *= 0.5*cos(a2[c]*out + b2[c]) + 0.5. f32 output.
// Block tile 256x64, wave tile 32x64 (same scheme as qkv_gemm).
// ---------------------------------------------------------------------------
__global__ __launch_bounds__(256) void proj_gemm(
    const __bf16* __restrict__ ab, const __bf16* __restrict__ wb,
    const float* __restrict__ a2, const float* __restrict__ b2,
    float* __restrict__ out) {
  const int lane = threadIdx.x & 31;
  const int wave = threadIdx.x >> 5;
  const int m0 = blockIdx.y * 256 + wave * 32;
  const int n0 = blockIdx.x * 64;
  const int lrow = lane & 15;
  const int half = lane >> 4;
  const int rowBase = half * 8;

  const __bf16* arow0 = ab + (size_t)(m0 + lrow) * CH;
  const __bf16* arow1 = ab + (size_t)(m0 + 16 + lrow) * CH;
  v8f acc0[4] = {};
  v8f acc1[4] = {};
  for (int kk = 0; kk < CH; kk += 32) {
    v16bf a0 = load_afrag(arow0 + kk, lane);
    v16bf a1 = load_afrag(arow1 + kk, lane);
    __builtin_prefetch(arow0 + kk + 256, 0, 3);
    __builtin_prefetch(arow1 + kk + 256, 0, 3);
#pragma unroll
    for (int nt = 0; nt < 4; ++nt) {
      const __bf16* brow = wb + (size_t)(n0 + nt * 16 + lrow) * CH + kk + half * 16;
      if (nt == 0) __builtin_prefetch(brow + 256, 0, 3);
      v16bf b = *(const v16bf*)brow;
      acc0[nt] = wmma_bf16(a0, b, acc0[nt]);
      acc1[nt] = wmma_bf16(a1, b, acc1[nt]);
    }
  }
#pragma unroll
  for (int g = 0; g < 2; ++g) {
#pragma unroll
    for (int nt = 0; nt < 4; ++nt) {
      int c = n0 + nt * 16 + lrow;
      float av = a2[c];
      float bv = b2[c];
#pragma unroll
      for (int r = 0; r < 8; ++r) {
        int m = m0 + g * 16 + rowBase + r;
        float z = g ? acc1[nt][r] : acc0[nt][r];
        out[(size_t)m * CH + c] = z * (0.5f * __cosf(av * z + bv) + 0.5f);
      }
    }
  }
}

// ---------------------------------------------------------------------------
// Launcher. Workspace layout (bytes):
//   xb     @ 0         (8 MB)   x as bf16
//   wqkvb  @ 8388608   (6 MB)
//   wprojb @ 14680064  (2 MB)
//   qb     @ 16777216  (8 MB)   (B,H,T,hs) bf16, pre-scaled
//   kb     @ 25165824  (8 MB)   (B,H,T,hs) bf16
//   vT     @ 33554432  (8 MB)   (B,H,hs,T) bf16
//   attn   @ 41943040  (8 MB)   (B,T,C)    bf16
//   mbuf   @ 50331648  (256 KB) row max
//   lbuf   @ 50593792  (256 KB) row sumexp
// ---------------------------------------------------------------------------
extern "C" void kernel_launch(void* const* d_in, const int* in_sizes, int n_in,
                              void* d_out, int out_size, void* d_ws, size_t ws_size,
                              hipStream_t stream) {
  (void)in_sizes; (void)n_in; (void)out_size; (void)ws_size;
  const float* x      = (const float*)d_in[0];
  const float* w_qkv  = (const float*)d_in[1];
  const float* w_proj = (const float*)d_in[2];
  const float* a1     = (const float*)d_in[3];
  const float* b1     = (const float*)d_in[4];
  const float* a2     = (const float*)d_in[5];
  const float* b2     = (const float*)d_in[6];
  float* out = (float*)d_out;

  char* ws = (char*)d_ws;
  __bf16* xb     = (__bf16*)(ws + 0);
  __bf16* wqkvb  = (__bf16*)(ws + 8388608);
  __bf16* wprojb = (__bf16*)(ws + 14680064);
  __bf16* qb     = (__bf16*)(ws + 16777216);
  __bf16* kb     = (__bf16*)(ws + 25165824);
  __bf16* vT     = (__bf16*)(ws + 33554432);
  __bf16* attn   = (__bf16*)(ws + 41943040);
  float*  mbuf   = (float*)(ws + 50331648);
  float*  lbuf   = (float*)(ws + 50593792);

  cvt_f32_bf16<<<1024, 256, 0, stream>>>(x, xb, BT * CH);
  cvt_f32_bf16<<<1024, 256, 0, stream>>>(w_qkv, wqkvb, 3 * CH * CH);
  cvt_f32_bf16<<<512, 256, 0, stream>>>(w_proj, wprojb, CH * CH);

  qkv_gemm<<<dim3(48, 16), 256, 0, stream>>>(xb, wqkvb, qb, kb, vT);
  attn_stats<<<dim3(BH, SEQ / 128), 256, 0, stream>>>(qb, kb, mbuf, lbuf);
  attn_pv<<<dim3(BH, SEQ / 128), 256, 0, stream>>>(qb, kb, vT, mbuf, lbuf,
                                                   a1, b1, attn);
  proj_gemm<<<dim3(16, 16), 256, 0, stream>>>(attn, wprojb, a2, b2, out);
}
